// PreT_Attention_463856468365
// MI455X (gfx1250) — matrix operand order repaired and checked by reading, not compiled, not run on hardware
//
#include <hip/hip_runtime.h>
#include <hip/hip_bf16.h>

// ---------------------------------------------------------------------------
// PreT attention for MI455X (gfx1250, wave32, WMMA bf16 16x16x32, f32 accum)
//   B=8, N=1024, C=768, H=12, D=64, P=64, M=P+N=1088
// Pipeline: f32->bf16 staging, QKV GEMM (WMMA, double-buffered LDS with
// async-to-LDS copies when available), flash attention (WMMA), output
// projection GEMM (WMMA) -> f32 out.
// ---------------------------------------------------------------------------

typedef __bf16 bf16;
typedef bf16 v16bf __attribute__((ext_vector_type(16)));
typedef bf16 v8bf  __attribute__((ext_vector_type(8)));
typedef float v8f  __attribute__((ext_vector_type(8)));
typedef int   v4i  __attribute__((ext_vector_type(4)));

#define BATCH  8
#define SEQ    1024
#define CH     768
#define NHEADS 12
#define HD     64
#define PLEN   64
#define MKV    1088   // PLEN + SEQ

#if defined(__has_builtin)
#if __has_builtin(__builtin_amdgcn_global_load_async_to_lds_b128) && \
    __has_builtin(__builtin_amdgcn_s_wait_asynccnt)
#define USE_ASYNC_COPY 1
#endif
#endif

__device__ __forceinline__ v16bf join8(v8bf lo, v8bf hi) {
  v16bf r;
#pragma unroll
  for (int i = 0; i < 8; ++i) { r[i] = lo[i]; r[i + 8] = hi[i]; }
  return r;
}

__device__ __forceinline__ v8f wmma_bf16(v16bf a, v16bf b, v8f c) {
  // D(16x16,f32) = A(16x32,bf16) x B(32x16,bf16) + C
  return __builtin_amdgcn_wmma_f32_16x16x32_bf16(false, a, false, b,
                                                 (short)0, c, false, false);
}

#if USE_ASYNC_COPY
// Builtin signature (from hipcc diagnostic): param0 = int4 in global (AS1),
// param1 = int4 in LDS (AS3), then imm offset, imm cpol.
typedef __attribute__((address_space(1))) v4i as1_v4i;
typedef __attribute__((address_space(3))) v4i as3_v4i;
// 16B global -> LDS async copy (ASYNCcnt-tracked, per-lane addresses)
__device__ __forceinline__ void cp_async16(bf16* lds, const bf16* g) {
  __builtin_amdgcn_global_load_async_to_lds_b128((as1_v4i*)g, (as3_v4i*)lds,
                                                 0, 0);
}
#endif

// ---------------------------------------------------------------------------
// f32 -> bf16 elementwise conversion
// ---------------------------------------------------------------------------
__global__ void cvt_f32_to_bf16(const float* __restrict__ in,
                                bf16* __restrict__ out, int n) {
  int i = blockIdx.x * blockDim.x + threadIdx.x;
  if (i < n) out[i] = (bf16)in[i];
}

// prompt (B,2,P,H,D) f32 -> K prefix [bh][key][d] and V^T prefix [bh][d][key]
__global__ void prompt_to_kv(const float* __restrict__ prompt,
                             bf16* __restrict__ kbuf, bf16* __restrict__ vtbuf) {
  int i = blockIdx.x * blockDim.x + threadIdx.x;
  if (i >= BATCH * 2 * PLEN * NHEADS * HD) return;
  int d = i % HD;     int t = i / HD;
  int h = t % NHEADS; t /= NHEADS;
  int p = t % PLEN;   t /= PLEN;
  int s = t & 1;      int b = t >> 1;
  bf16 v = (bf16)prompt[i];
  if (s == 0)
    kbuf[(((size_t)b * NHEADS + h) * MKV + p) * HD + d] = v;
  else
    vtbuf[(((size_t)b * NHEADS + h) * HD + d) * MKV + p] = v;
}

// ---------------------------------------------------------------------------
// Shared GEMM mainloop: C[128x64] tile of X[M,K] @ W[N,K]^T, bf16, f32 acc.
// Block = 256 threads = 8 waves, wave grid 4(M) x 2(N), 32x32 per wave.
// Double-buffered LDS; rows padded to 40 bf16 (80B) for bank spread.
// Async path: issue tile t+1 into the idle buffer while WMMAs consume tile t;
//             one s_wait_asynccnt + one barrier per K-step.
// ---------------------------------------------------------------------------
#define A_STRIDE (128 * 40)
#define B_STRIDE (64 * 40)

template <int KDIM>
__device__ __forceinline__ void gemm_mainloop(const bf16* __restrict__ X,
                                              const bf16* __restrict__ W,
                                              int mb, int nb,
                                              bf16* As, bf16* Bs, v8f acc[2][2]) {
  const int tid  = threadIdx.x;
  const int lane = tid & 31;
  const int w    = tid >> 5;
  const int g    = lane >> 4;     // half-wave group (0/1)
  const int ln   = lane & 15;
  const int wm   = (w & 3) * 32;
  const int wn   = (w >> 2) * 32;
  const int row  = tid >> 2;      // 0..63
  const int cc   = tid & 3;       // 16B chunk within 32-wide K slab

  auto compute = [&](int buf) {
    const bf16* Ab = As + buf * A_STRIDE;
    const bf16* Bb = Bs + buf * B_STRIDE;
    v16bf a[2], bb[2];
#pragma unroll
    for (int ti = 0; ti < 2; ++ti) {
      // A frag: lane holds row (L%16), K = 16g..16g+15 (contiguous 32B)
      const bf16* p = Ab + (wm + 16 * ti + ln) * 40;
      a[ti] = join8(*(const v8bf*)(p + 16 * g), *(const v8bf*)(p + 16 * g + 8));
    }
#pragma unroll
    for (int tj = 0; tj < 2; ++tj) {
      // B frag: lane holds col (L%16), K = 16g..16g+15 (contiguous 32B)
      const bf16* p = Bb + (wn + 16 * tj + ln) * 40 + 16 * g;
      bb[tj] = join8(*(const v8bf*)p, *(const v8bf*)(p + 8));
    }
#pragma unroll
    for (int ti = 0; ti < 2; ++ti)
#pragma unroll
      for (int tj = 0; tj < 2; ++tj)
        acc[ti][tj] = wmma_bf16(a[ti], bb[tj], acc[ti][tj]);
  };

#if USE_ASYNC_COPY
  auto issue = [&](int kb, int buf) {
    cp_async16(As + buf * A_STRIDE + row * 40 + cc * 8,
               X + (size_t)(mb + row) * KDIM + kb + cc * 8);
    cp_async16(As + buf * A_STRIDE + (row + 64) * 40 + cc * 8,
               X + (size_t)(mb + row + 64) * KDIM + kb + cc * 8);
    cp_async16(Bs + buf * B_STRIDE + row * 40 + cc * 8,
               W + (size_t)(nb + row) * KDIM + kb + cc * 8);
  };
  issue(0, 0);
  for (int kb = 0; kb < KDIM; kb += 32) {
    const int buf = (kb >> 5) & 1;
    __builtin_amdgcn_s_wait_asynccnt(0);  // my slice of buf landed
    __syncthreads();                       // everyone's slice landed
    if (kb + 32 < KDIM) issue(kb + 32, buf ^ 1);
    compute(buf);
    // no trailing barrier: next overwrite of `buf` is two iterations away,
    // and the WMMA operand waits retire this iteration's ds_loads before
    // the next barrier is passed.
  }
#else
  // Fallback: register-staged double buffer (global loads for tile t+1
  // issued before compute of tile t).
  v8bf ra0, ra1, rb0;
  auto gload = [&](int kb) {
    ra0 = *(const v8bf*)(X + (size_t)(mb + row) * KDIM + kb + cc * 8);
    ra1 = *(const v8bf*)(X + (size_t)(mb + row + 64) * KDIM + kb + cc * 8);
    rb0 = *(const v8bf*)(W + (size_t)(nb + row) * KDIM + kb + cc * 8);
  };
  auto sstore = [&](int buf) {
    *(v8bf*)(As + buf * A_STRIDE + row * 40 + cc * 8) = ra0;
    *(v8bf*)(As + buf * A_STRIDE + (row + 64) * 40 + cc * 8) = ra1;
    *(v8bf*)(Bs + buf * B_STRIDE + row * 40 + cc * 8) = rb0;
  };
  gload(0);
  sstore(0);
  for (int kb = 0; kb < KDIM; kb += 32) {
    const int buf = (kb >> 5) & 1;
    __syncthreads();                       // buf writes visible
    if (kb + 32 < KDIM) gload(kb + 32);    // overlap with compute below
    compute(buf);
    if (kb + 32 < KDIM) {
      __syncthreads();                     // all reads of buf^1 done
      sstore(buf ^ 1);
    }
  }
#endif
}

// ---------------------------------------------------------------------------
// Kernel 1: qkv = x @ W_qkv^T ; epilogue scatters into q/k/v^T layouts (bf16)
// grid = (8192/128) x (2304/64) = 64*36
// ---------------------------------------------------------------------------
__global__ __launch_bounds__(256) void gemm_qkv(const bf16* __restrict__ X,
                                                const bf16* __restrict__ W,
                                                bf16* __restrict__ qo,
                                                bf16* __restrict__ ko,
                                                bf16* __restrict__ vto) {
  __shared__ __align__(16) bf16 As[2 * A_STRIDE];
  __shared__ __align__(16) bf16 Bs[2 * B_STRIDE];
  const int mb = (blockIdx.x & 63) * 128;
  const int nb = (blockIdx.x >> 6) * 64;
  v8f acc[2][2] = {};
  gemm_mainloop<CH>(X, W, mb, nb, As, Bs, acc);

  const int lane = threadIdx.x & 31, w = threadIdx.x >> 5;
  const int g = lane >> 4, ln = lane & 15;
  const int wm = (w & 3) * 32, wn = (w >> 2) * 32;
#pragma unroll
  for (int ti = 0; ti < 2; ++ti)
#pragma unroll
    for (int tj = 0; tj < 2; ++tj)
#pragma unroll
      for (int r = 0; r < 8; ++r) {
        int m = mb + wm + 16 * ti + r + 8 * g;   // row in [0, B*N)
        int n = nb + wn + 16 * tj + ln;          // col in [0, 3C)
        int b = m >> 10, nrow = m & 1023;
        bf16 v = (bf16)acc[ti][tj][r];
        if (n < CH) {                                   // Q: [bh][n][d]
          int h = n >> 6, d = n & 63;
          qo[(((size_t)b * NHEADS + h) * SEQ + nrow) * HD + d] = v;
        } else if (n < 2 * CH) {                        // K: [bh][P+n][d]
          int j = n - CH, h = j >> 6, d = j & 63;
          ko[(((size_t)b * NHEADS + h) * MKV + PLEN + nrow) * HD + d] = v;
        } else {                                        // V^T: [bh][d][P+n]
          int j = n - 2 * CH, h = j >> 6, d = j & 63;
          vto[(((size_t)b * NHEADS + h) * HD + d) * MKV + PLEN + nrow] = v;
        }
      }
}

// ---------------------------------------------------------------------------
// Kernel 2: flash attention. Block = 128 thr = 4 waves; wave owns 16 q-rows.
// grid = B*H*(N/64) = 1536. 34 chunks of 32 keys (M=1088).
// ---------------------------------------------------------------------------
__global__ __launch_bounds__(128) void attn_flash(const bf16* __restrict__ qb,
                                                  const bf16* __restrict__ kbuf,
                                                  const bf16* __restrict__ vtbuf,
                                                  bf16* __restrict__ aob) {
  __shared__ __align__(16) bf16 Pb[4][16 * 40];   // per-wave P buffer (padded)
  const int bh    = blockIdx.x >> 4;              // b*H + h
  const int qtile = blockIdx.x & 15;
  const int b = bh / NHEADS, h = bh % NHEADS;
  const int wv = threadIdx.x >> 5, lane = threadIdx.x & 31;
  const int g = lane >> 4, ln = lane & 15;
  bf16* P = Pb[wv];

  // Q A-frags (K = d, two 32-wide K-steps), loaded once
  const bf16* qrowp = qb + ((size_t)bh * SEQ + qtile * 64 + wv * 16 + ln) * HD;
  v16bf qa0 = join8(*(const v8bf*)(qrowp + 16 * g),      *(const v8bf*)(qrowp + 8 + 16 * g));
  v16bf qa1 = join8(*(const v8bf*)(qrowp + 32 + 16 * g), *(const v8bf*)(qrowp + 40 + 16 * g));

  const bf16* kbase  = kbuf  + (size_t)bh * MKV * HD;
  const bf16* vtbase = vtbuf + (size_t)bh * HD * MKV;

  float mi[8], li[8];
  v8f acc[4] = {};
#pragma unroll
  for (int r = 0; r < 8; ++r) { mi[r] = -1e30f; li[r] = 0.f; }

  for (int kc = 0; kc < MKV / 32; ++kc) {
    const int key0 = kc * 32;
    // K^T B-frags: lane = key column, K(=d) contiguous in k row
    const bf16* kr0 = kbase + (size_t)(key0 + ln) * HD + 16 * g;
    const bf16* kr1 = kr0 + (size_t)16 * HD;
    v16bf b00 = join8(*(const v8bf*)(kr0),      *(const v8bf*)(kr0 + 8));
    v16bf b01 = join8(*(const v8bf*)(kr0 + 32), *(const v8bf*)(kr0 + 40));
    v16bf b10 = join8(*(const v8bf*)(kr1),      *(const v8bf*)(kr1 + 8));
    v16bf b11 = join8(*(const v8bf*)(kr1 + 32), *(const v8bf*)(kr1 + 40));

    v8f s0 = {}, s1 = {};
    s0 = wmma_bf16(qa0, b00, s0);
    s0 = wmma_bf16(qa1, b01, s0);
    s1 = wmma_bf16(qa0, b10, s1);
    s1 = wmma_bf16(qa1, b11, s1);

    // online softmax over the 32-key chunk (row = r + 8g, across 16 lanes)
    float rmax[8];
#pragma unroll
    for (int r = 0; r < 8; ++r) {
      s0[r] *= 0.125f;  // D^-0.5
      s1[r] *= 0.125f;
      rmax[r] = fmaxf(s0[r], s1[r]);
    }
#pragma unroll
    for (int off = 1; off < 16; off <<= 1)
#pragma unroll
      for (int r = 0; r < 8; ++r)
        rmax[r] = fmaxf(rmax[r], __shfl_xor(rmax[r], off, 32));

    float alpha[8], rsum[8];
#pragma unroll
    for (int r = 0; r < 8; ++r) {
      float nm = fmaxf(mi[r], rmax[r]);
      alpha[r] = __expf(mi[r] - nm);
      mi[r] = nm;
      float p0 = __expf(s0[r] - nm);
      float p1 = __expf(s1[r] - nm);
      s0[r] = p0; s1[r] = p1;
      rsum[r] = p0 + p1;
    }
#pragma unroll
    for (int off = 1; off < 16; off <<= 1)
#pragma unroll
      for (int r = 0; r < 8; ++r)
        rsum[r] += __shfl_xor(rsum[r], off, 32);
#pragma unroll
    for (int r = 0; r < 8; ++r) li[r] = li[r] * alpha[r] + rsum[r];
#pragma unroll
    for (int t = 0; t < 4; ++t)
#pragma unroll
      for (int r = 0; r < 8; ++r) acc[t][r] *= alpha[r];

    // C-layout -> A-layout for P via LDS bounce
#pragma unroll
    for (int r = 0; r < 8; ++r) {
      P[(r + 8 * g) * 40 + ln]      = (bf16)s0[r];
      P[(r + 8 * g) * 40 + 16 + ln] = (bf16)s1[r];
    }
    __syncthreads();
    const bf16* pr = P + ln * 40;
    v16bf pa = join8(*(const v8bf*)(pr + 16 * g), *(const v8bf*)(pr + 16 * g + 8));

    // P @ V : B-frags from V^T rows (keys contiguous)
#pragma unroll
    for (int t = 0; t < 4; ++t) {
      const bf16* vr = vtbase + (size_t)(16 * t + ln) * MKV + key0 + 16 * g;
      v16bf vb = join8(*(const v8bf*)(vr), *(const v8bf*)(vr + 8));
      acc[t] = wmma_bf16(pa, vb, acc[t]);
    }
    __syncthreads();
  }

  // normalize + write attention output as (B, N, C) bf16
#pragma unroll
  for (int t = 0; t < 4; ++t)
#pragma unroll
    for (int r = 0; r < 8; ++r) {
      int m = r + 8 * g;
      int nrow = qtile * 64 + wv * 16 + m;
      int d = 16 * t + ln;
      float v = acc[t][r] / li[r];
      aob[((size_t)b * SEQ + nrow) * CH + h * HD + d] = (bf16)v;
    }
}

// ---------------------------------------------------------------------------
// Kernel 3: out = ao @ W_proj^T + b_proj   (f32 output)
// grid = 64 * (768/64) = 64*12
// ---------------------------------------------------------------------------
__global__ __launch_bounds__(256) void gemm_proj(const bf16* __restrict__ A,
                                                 const bf16* __restrict__ W,
                                                 const float* __restrict__ bias,
                                                 float* __restrict__ out) {
  __shared__ __align__(16) bf16 As[2 * A_STRIDE];
  __shared__ __align__(16) bf16 Bs[2 * B_STRIDE];
  const int mb = (blockIdx.x & 63) * 128;
  const int nb = (blockIdx.x >> 6) * 64;
  v8f acc[2][2] = {};
  gemm_mainloop<CH>(A, W, mb, nb, As, Bs, acc);

  const int lane = threadIdx.x & 31, w = threadIdx.x >> 5;
  const int g = lane >> 4, ln = lane & 15;
  const int wm = (w & 3) * 32, wn = (w >> 2) * 32;
#pragma unroll
  for (int ti = 0; ti < 2; ++ti)
#pragma unroll
    for (int tj = 0; tj < 2; ++tj)
#pragma unroll
      for (int r = 0; r < 8; ++r) {
        int m = mb + wm + 16 * ti + r + 8 * g;
        int n = nb + wn + 16 * tj + ln;
        out[(size_t)m * CH + n] = acc[ti][tj][r] + bias[n];
      }
}

// ---------------------------------------------------------------------------
extern "C" void kernel_launch(void* const* d_in, const int* in_sizes, int n_in,
                              void* d_out, int out_size, void* d_ws, size_t ws_size,
                              hipStream_t stream) {
  const float* x      = (const float*)d_in[0];
  const float* prompt = (const float*)d_in[1];
  const float* W_qkv  = (const float*)d_in[2];
  const float* W_proj = (const float*)d_in[3];
  const float* b_proj = (const float*)d_in[4];
  float* out = (float*)d_out;

  // workspace carve-up (bf16 staging; ~66 MB total, 256B-aligned slabs)
  char* ws = (char*)d_ws;
  size_t off = 0;
  auto alloc = [&](size_t bytes) {
    char* p = ws + off;
    off += (bytes + 255) & ~(size_t)255;
    return p;
  };
  bf16* x_bf    = (bf16*)alloc((size_t)BATCH * SEQ * CH * 2);
  bf16* wqkv_bf = (bf16*)alloc((size_t)3 * CH * CH * 2);
  bf16* wproj_bf= (bf16*)alloc((size_t)CH * CH * 2);
  bf16* q_bf    = (bf16*)alloc((size_t)BATCH * NHEADS * SEQ * HD * 2);
  bf16* k_bf    = (bf16*)alloc((size_t)BATCH * NHEADS * MKV * HD * 2);
  bf16* vt_bf   = (bf16*)alloc((size_t)BATCH * NHEADS * HD * MKV * 2);
  bf16* ao_bf   = (bf16*)alloc((size_t)BATCH * SEQ * CH * 2);

  const int nx  = BATCH * SEQ * CH;
  const int nwq = 3 * CH * CH;
  const int nwp = CH * CH;
  const int np  = BATCH * 2 * PLEN * NHEADS * HD;
  cvt_f32_to_bf16<<<(nx  + 255) / 256, 256, 0, stream>>>(x, x_bf, nx);
  cvt_f32_to_bf16<<<(nwq + 255) / 256, 256, 0, stream>>>(W_qkv, wqkv_bf, nwq);
  cvt_f32_to_bf16<<<(nwp + 255) / 256, 256, 0, stream>>>(W_proj, wproj_bf, nwp);
  prompt_to_kv<<<(np + 255) / 256, 256, 0, stream>>>(prompt, k_bf, vt_bf);

  gemm_qkv<<<64 * 36, 256, 0, stream>>>(x_bf, wqkv_bf, q_bf, k_bf, vt_bf);
  attn_flash<<<BATCH * NHEADS * (SEQ / 64), 128, 0, stream>>>(q_bf, k_bf, vt_bf, ao_bf);
  gemm_proj<<<64 * 12, 256, 0, stream>>>(ao_bf, wproj_bf, b_proj, out);
}
